// MemoryAttentionNetwork_42898133353132
// MI455X (gfx1250) — matrix operand order repaired and checked by reading, not compiled, not run on hardware
//
#include <hip/hip_runtime.h>
#include <hip/hip_bf16.h>

// ---------------- problem constants ----------------
#define DEPTH_ 12
#define BATCH_ 4
#define LMEM_ 256
#define MEMLEN_ 2048
#define SEQ_ 2048
#define DIM_ 512
#define HEADS_ 8
#define DH_ 64
#define MB_ (DEPTH_ * BATCH_)          // 48
#define QROWS_ (MB_ * LMEM_)           // 12288
#define NK_ (LMEM_ + MEMLEN_ + SEQ_)   // 4352
#define NMEM_ELEMS ((size_t)DEPTH_ * BATCH_ * MEMLEN_ * DIM_)  // 50331648
#define SCALE_ 0.35355339059327373f    // 64^-0.25
#define XPITCH (DIM_ + 8)              // 520 (1040 B, 16B multiple)
#define TPITCH 40                      // transposed LDS pitch (80 B, 16B multiple)

typedef __attribute__((ext_vector_type(16))) __bf16 bf16x16;
typedef __attribute__((ext_vector_type(8))) float f32x8;
typedef __attribute__((ext_vector_type(8))) unsigned short u16x8;

__device__ __forceinline__ unsigned short f2bf(float f) {
  unsigned int u = __float_as_uint(f);
  u += 0x7FFFu + ((u >> 16) & 1u);   // round-to-nearest-even
  return (unsigned short)(u >> 16);
}

struct FragU { union { bf16x16 v; unsigned short u[16]; u16x8 h[2]; }; };
union Pack8 { u16x8 v; unsigned short u[8]; };

__device__ __forceinline__ f32x8 zero8() {
  f32x8 z = {0.f, 0.f, 0.f, 0.f, 0.f, 0.f, 0.f, 0.f};
  return z;
}

__device__ __forceinline__ f32x8 wmma_bf16(bf16x16 a, bf16x16 b, f32x8 c) {
  return __builtin_amdgcn_wmma_f32_16x16x32_bf16(false, a, false, b, (short)0, c, false, false);
}

// A fragment (16x32) from row-major [M][K] storage. Lane r(0..15)=row,
// lane group g selects K runs {g*8..g*8+7} and {16+g*8..16+g*8+7}
// (ISA 7.12.2) -> two 16-byte loads.
__device__ __forceinline__ bf16x16 load_a_vec(const unsigned short* base, int ld,
                                              int mrow, int kcol, int lane) {
  int g = lane >> 4, r = lane & 15;
  const unsigned short* p = base + (size_t)(mrow + r) * ld + kcol + g * 8;
  FragU f;
  f.h[0] = *(const u16x8*)p;
  f.h[1] = *(const u16x8*)(p + 16);
  return f.v;
}

// B fragment (32x16) from N-major [N][K] storage (i.e. transposed weights).
// Lane c = N column, group g: K runs {krow+g*8..+7} and {krow+16+g*8..+7} -> two 16-byte loads.
__device__ __forceinline__ bf16x16 load_bT_vec(const unsigned short* base, int ld,
                                               int krow, int ncol, int lane) {
  int g = lane >> 4, c = lane & 15;
  const unsigned short* p = base + (size_t)(ncol + c) * ld + krow + g * 8;
  FragU f;
  f.h[0] = *(const u16x8*)p;
  f.h[1] = *(const u16x8*)(p + 16);
  return f.v;
}

// ---------------- trivial kernels ----------------
__global__ void k_zero(float* __restrict__ p, int n) {
  int s = gridDim.x * blockDim.x;
  for (int i = blockIdx.x * blockDim.x + threadIdx.x; i < n; i += s) p[i] = 0.f;
}

__global__ void k_copy4(const float4* __restrict__ src, float4* __restrict__ dst, int n4) {
  int s = gridDim.x * blockDim.x;
  for (int i = blockIdx.x * blockDim.x + threadIdx.x; i < n4; i += s) dst[i] = src[i];
}

// Weights -> bf16, stored TRANSPOSED ([out][in]) so B fragments are two
// contiguous 16-byte loads.
__global__ void k_wconv(const float* __restrict__ wq, const float* __restrict__ wkv,
                        const float* __restrict__ wout,
                        unsigned short* __restrict__ wqbT, unsigned short* __restrict__ wkbT,
                        unsigned short* __restrict__ wvbT, unsigned short* __restrict__ wobT) {
  int s = gridDim.x * blockDim.x;
  for (int i = blockIdx.x * blockDim.x + threadIdx.x; i < DIM_ * DIM_; i += s) {
    int d = i >> 9, e = i & (DIM_ - 1);
    wqbT[e * DIM_ + d] = f2bf(wq[i]);
    wobT[e * DIM_ + d] = f2bf(wout[i]);
  }
  for (int i = blockIdx.x * blockDim.x + threadIdx.x; i < DIM_ * 2 * DIM_; i += s) {
    int d = i >> 10, e = i & 1023;  // w_kv is (512, 1024) row-major
    unsigned short v = f2bf(wkv[i]);
    if (e < DIM_) wkbT[e * DIM_ + d] = v;
    else          wvbT[(e - DIM_) * DIM_ + d] = v;
  }
}

// LayerNorm over last dim (512), no affine. One wave per row, 16 elems/lane.
__global__ void __launch_bounds__(256)
k_layernorm(const float* __restrict__ lmem, unsigned short* __restrict__ normed) {
  int wave = threadIdx.x >> 5, lane = threadIdx.x & 31;
  int row = blockIdx.x * 8 + wave;
  const float4* x4 = (const float4*)(lmem + (size_t)row * DIM_ + lane * 16);
  float v[16];
#pragma unroll
  for (int i = 0; i < 4; ++i) {
    float4 t = x4[i];
    v[4 * i] = t.x; v[4 * i + 1] = t.y; v[4 * i + 2] = t.z; v[4 * i + 3] = t.w;
  }
  float s = 0.f, q = 0.f;
#pragma unroll
  for (int i = 0; i < 16; ++i) { s += v[i]; q += v[i] * v[i]; }
#pragma unroll
  for (int off = 16; off >= 1; off >>= 1) {
    s += __shfl_xor(s, off, 32);
    q += __shfl_xor(q, off, 32);
  }
  float mu = s * (1.f / DIM_);
  float var = q * (1.f / DIM_) - mu * mu;
  float rs = rsqrtf(var + 1e-5f);
  unsigned short* o = normed + (size_t)row * DIM_ + lane * 16;
  Pack8 p0, p1;
#pragma unroll
  for (int i = 0; i < 8; ++i) {
    p0.u[i] = f2bf((v[i] - mu) * rs);
    p1.u[i] = f2bf((v[8 + i] - mu) * rs);
  }
  *(u16x8*)o = p0.v;
  *(u16x8*)(o + 8) = p1.v;
}

// ---------------- fused K/V projection + exp + context accumulation ----------------
// grid = (17 chunks, 48 mb); 256 threads = 8 waves; wave == head for the
// context GEMM. Each chunk covers 256 rows of nk (8 tiles of 32 rows).
// Ek and V are kept TRANSPOSED in LDS ([d][n], pitch 40) so both the A^T and
// the B fragment of the context GEMM are pairs of 16-byte ds loads.
__global__ void __launch_bounds__(256)
k_context(const float* __restrict__ smem, const float* __restrict__ hiddens,
          const unsigned short* __restrict__ normed,
          const unsigned short* __restrict__ wkbT, const unsigned short* __restrict__ wvbT,
          float* __restrict__ ctx_num, float* __restrict__ denomG) {
  __shared__ __align__(16) unsigned short Xs[32][XPITCH];
  __shared__ __align__(16) unsigned short Ekt[DIM_][TPITCH];
  __shared__ __align__(16) unsigned short Vt[DIM_][TPITCH];
  __shared__ float denom_s[DIM_];

  const int tid = threadIdx.x;
  const int wave = tid >> 5, lane = tid & 31;
  const int g = lane >> 4, cl = lane & 15;
  const int chunk = blockIdx.x;  // 0..16
  const int mb = blockIdx.y;     // 0..47

  for (int i = tid; i < DIM_; i += 256) denom_s[i] = 0.f;

  f32x8 cacc[4][4];
#pragma unroll
  for (int a = 0; a < 4; ++a)
#pragma unroll
    for (int b = 0; b < 4; ++b) cacc[a][b] = zero8();

  for (int it = 0; it < 8; ++it) {
    int nbase = chunk * 256 + it * 32;
    __syncthreads();
    // Stage 32x512 bf16 tile of kv_input = concat(normed_lmem, smem, hiddens),
    // 8 elements (16 B) per thread-iteration.
    for (int t = tid; t < 32 * DIM_ / 8; t += 256) {
      int rr = t >> 6;              // (t*8)/512
      int cc = (t & 63) * 8;        // (t*8)%512
      int n = nbase + rr;
      Pack8 pk;
      if (n < LMEM_) {
        pk.v = *(const u16x8*)(normed + ((size_t)mb * LMEM_ + n) * DIM_ + cc);
      } else {
        const float* src = (n < LMEM_ + MEMLEN_)
            ? smem + ((size_t)mb * MEMLEN_ + (n - LMEM_)) * DIM_ + cc
            : hiddens + ((size_t)mb * SEQ_ + (n - LMEM_ - MEMLEN_)) * DIM_ + cc;
        float4 a = ((const float4*)src)[0];
        float4 b = ((const float4*)src)[1];
        pk.u[0] = f2bf(a.x); pk.u[1] = f2bf(a.y); pk.u[2] = f2bf(a.z); pk.u[3] = f2bf(a.w);
        pk.u[4] = f2bf(b.x); pk.u[5] = f2bf(b.y); pk.u[6] = f2bf(b.z); pk.u[7] = f2bf(b.w);
        __builtin_prefetch((const void*)(src + 32 * DIM_), 0, 1);  // next tile
      }
      *(u16x8*)(&Xs[rr][cc]) = pk.v;
    }
    __syncthreads();
    // K = X@Wk (then exp(scale*K)) and V = X@Wv; results stored transposed.
    for (int i = 0; i < 8; ++i) {
      int t = wave * 8 + i;
      int mt = t & 1, nt = t >> 1;
      f32x8 ak = zero8(), av = zero8();
      for (int kk = 0; kk < DIM_; kk += 32) {
        bf16x16 a = load_a_vec(&Xs[0][0], XPITCH, mt * 16, kk, lane);
        bf16x16 bk = load_bT_vec(wkbT, DIM_, kk, nt * 16, lane);
        ak = wmma_bf16(a, bk, ak);
        bf16x16 bv = load_bT_vec(wvbT, DIM_, kk, nt * 16, lane);
        av = wmma_bf16(a, bv, av);
      }
      // D-fragment: value j lives at (n = mt*16 + j + 8g, d = nt*16 + cl).
      // Transposed store: one 16-byte run along n per lane.
      Pack8 pe, pv;
      float sums = 0.f;
#pragma unroll
      for (int j = 0; j < 8; ++j) {
        float e = __expf(SCALE_ * ak[j]);
        pe.u[j] = f2bf(e);
        pv.u[j] = f2bf(av[j]);
        sums += e;
      }
      int d = nt * 16 + cl;
      int nb = mt * 16 + 8 * g;
      *(u16x8*)(&Ekt[d][nb]) = pe.v;
      *(u16x8*)(&Vt[d][nb]) = pv.v;
      atomicAdd(&denom_s[d], sums);
    }
    __syncthreads();
    // context[h] += Ek^T @ V over these 32 rows (wave == head).
    const int h = wave;
#pragma unroll
    for (int mt = 0; mt < 4; ++mt) {
      bf16x16 a = load_a_vec(&Ekt[0][0], TPITCH, h * DH_ + mt * 16, 0, lane);
#pragma unroll
      for (int nt = 0; nt < 4; ++nt) {
        bf16x16 b = load_bT_vec(&Vt[0][0], TPITCH, 0, h * DH_ + nt * 16, lane);
        cacc[mt][nt] = wmma_bf16(a, b, cacc[mt][nt]);
      }
    }
  }
  // Flush per-chunk partials with f32 atomics (L2-resident 6.3 MB region).
  {
    const int h = wave;
    float* ctxp = ctx_num + (size_t)(mb * HEADS_ + h) * DH_ * DH_;
#pragma unroll
    for (int mt = 0; mt < 4; ++mt)
#pragma unroll
      for (int nt = 0; nt < 4; ++nt)
#pragma unroll
        for (int j = 0; j < 8; ++j)
          atomicAdd(&ctxp[(mt * 16 + j + 8 * g) * DH_ + nt * 16 + cl], cacc[mt][nt][j]);
  }
  __syncthreads();
  for (int i = tid; i < DIM_; i += 256) atomicAdd(&denomG[mb * DIM_ + i], denom_s[i]);
}

// context = num / denom, stored bf16 TRANSPOSED ([mb,h,e,d]) for vectorized
// B fragments in the q@context GEMM.
__global__ void k_ctxfin(const float* __restrict__ ctx_num, const float* __restrict__ denomG,
                         unsigned short* __restrict__ ctxbfT) {
  const int total = MB_ * HEADS_ * DH_ * DH_;
  int s = gridDim.x * blockDim.x;
  for (int i = blockIdx.x * blockDim.x + threadIdx.x; i < total; i += s) {
    int mbh = i >> 12;           // mb*8 + h
    int d = (i >> 6) & 63;
    int e = i & 63;
    int hd = (i >> 6) & 511;     // h*64 + d
    float v = ctx_num[i] / denomG[(i >> 15) * DIM_ + hd];
    ctxbfT[(size_t)mbh * (DH_ * DH_) + e * DH_ + d] = f2bf(v);
  }
}

// ---------------- fused q-proj -> feature softmax -> q@context -> w_out + residual ----------------
// 32 lmem rows per block; 256 threads = 8 waves (wave == head for q@context).
__global__ void __launch_bounds__(256)
k_qpath(const float* __restrict__ lmem, const unsigned short* __restrict__ normed,
        const unsigned short* __restrict__ wqbT, const unsigned short* __restrict__ wobT,
        const unsigned short* __restrict__ ctxbfT, const float* __restrict__ rezero,
        float* __restrict__ nlm) {
  __shared__ __align__(16) unsigned short A16[32][XPITCH];  // Xn -> Qs(bf16) -> O(bf16)
  __shared__ __align__(16) float F32[32][DIM_ + 4];         // Q(f32) -> O(f32)
  const int tid = threadIdx.x;
  const int wave = tid >> 5, lane = tid & 31;
  const int g = lane >> 4, cl = lane & 15;
  const int r0 = blockIdx.x * 32;
  const int mb = r0 / LMEM_;

  // 1: stage normed lmem rows (already bf16), 16 B per thread-iteration.
  for (int t = tid; t < 32 * DIM_ / 8; t += 256) {
    int rr = t >> 6, cc = (t & 63) * 8;
    *(u16x8*)(&A16[rr][cc]) = *(const u16x8*)(normed + (size_t)(r0 + rr) * DIM_ + cc);
  }
  __syncthreads();
  // 2: Q = scale * (Xn @ Wq)
  for (int i = 0; i < 8; ++i) {
    int t = wave * 8 + i;
    int mt = t & 1, nt = t >> 1;
    f32x8 acc = zero8();
    for (int kk = 0; kk < DIM_; kk += 32) {
      bf16x16 a = load_a_vec(&A16[0][0], XPITCH, mt * 16, kk, lane);
      bf16x16 b = load_bT_vec(wqbT, DIM_, kk, nt * 16, lane);
      acc = wmma_bf16(a, b, acc);
    }
#pragma unroll
    for (int j = 0; j < 8; ++j)
      F32[mt * 16 + j + 8 * g][nt * 16 + cl] = SCALE_ * acc[j];
  }
  __syncthreads();
  // 3: softmax over the 64 features of each (row, head); one thread each.
  {
    int row = tid >> 3, h = tid & 7;
    float mx = -1e30f;
    for (int i = 0; i < 64; ++i) mx = fmaxf(mx, F32[row][h * 64 + i]);
    float sum = 0.f;
    for (int i = 0; i < 64; ++i) sum += __expf(F32[row][h * 64 + i] - mx);
    float inv = 1.f / sum;
    for (int i0 = 0; i0 < 64; i0 += 8) {
      Pack8 pk;
#pragma unroll
      for (int j = 0; j < 8; ++j)
        pk.u[j] = f2bf(__expf(F32[row][h * 64 + i0 + j] - mx) * inv);
      *(u16x8*)(&A16[row][h * 64 + i0]) = pk.v;
    }
  }
  __syncthreads();
  // 4: O[:, h] = Qs[:, h] @ context[h]   (wave == head)
  {
    int h = wave;
    const unsigned short* cbase = ctxbfT + (size_t)(mb * HEADS_ + h) * DH_ * DH_;
    for (int i = 0; i < 8; ++i) {
      int mt = i & 1, nt = i >> 1;  // 2 row-tiles x 4 col-tiles of 16
      f32x8 acc = zero8();
      for (int kk = 0; kk < DH_; kk += 32) {
        bf16x16 a = load_a_vec(&A16[0][0], XPITCH, mt * 16, h * DH_ + kk, lane);
        bf16x16 b = load_bT_vec(cbase, DH_, kk, nt * 16, lane);
        acc = wmma_bf16(a, b, acc);
      }
#pragma unroll
      for (int j = 0; j < 8; ++j)
        F32[mt * 16 + j + 8 * g][h * DH_ + nt * 16 + cl] = acc[j];
    }
  }
  __syncthreads();
  // 5: O f32 -> bf16 (8 elems / iteration)
  for (int t = tid; t < 32 * DIM_ / 8; t += 256) {
    int rr = t >> 6, cc = (t & 63) * 8;
    const float4* s4 = (const float4*)(&F32[rr][cc]);
    float4 a = s4[0], b = s4[1];
    Pack8 pk;
    pk.u[0] = f2bf(a.x); pk.u[1] = f2bf(a.y); pk.u[2] = f2bf(a.z); pk.u[3] = f2bf(a.w);
    pk.u[4] = f2bf(b.x); pk.u[5] = f2bf(b.y); pk.u[6] = f2bf(b.z); pk.u[7] = f2bf(b.w);
    *(u16x8*)(&A16[rr][cc]) = pk.v;
  }
  __syncthreads();
  // 6: next_lmem = g * (O @ Wout) + lmem
  float gz = rezero[0];
  for (int i = 0; i < 8; ++i) {
    int t = wave * 8 + i;
    int mt = t & 1, nt = t >> 1;
    f32x8 acc = zero8();
    for (int kk = 0; kk < DIM_; kk += 32) {
      bf16x16 a = load_a_vec(&A16[0][0], XPITCH, mt * 16, kk, lane);
      bf16x16 b = load_bT_vec(wobT, DIM_, kk, nt * 16, lane);
      acc = wmma_bf16(a, b, acc);
    }
#pragma unroll
    for (int j = 0; j < 8; ++j) {
      size_t idx = (size_t)(r0 + mt * 16 + j + 8 * g) * DIM_ + nt * 16 + cl;
      nlm[idx] = gz * acc[j] + lmem[idx];
    }
  }
}

// ---------------- host launcher ----------------
extern "C" void kernel_launch(void* const* d_in, const int* in_sizes, int n_in,
                              void* d_out, int out_size, void* d_ws, size_t ws_size,
                              hipStream_t stream) {
  (void)in_sizes; (void)n_in; (void)out_size; (void)ws_size;
  const float* lmem = (const float*)d_in[0];
  const float* smem = (const float*)d_in[1];
  const float* hiddens = (const float*)d_in[2];
  // d_in[3] = init_lmem (unused: lmem is non-empty)
  const float* w_q = (const float*)d_in[4];
  const float* w_kv = (const float*)d_in[5];
  const float* w_out = (const float*)d_in[6];
  const float* rezero = (const float*)d_in[7];

  float* nmem = (float*)d_out;
  float* nlm = nmem + NMEM_ELEMS;

  char* ws = (char*)d_ws;
  size_t off = 0;
  auto alloc = [&](size_t bytes) {
    void* p = ws + off;
    off = (off + bytes + 255) & ~(size_t)255;
    return p;
  };
  unsigned short* normed = (unsigned short*)alloc((size_t)QROWS_ * DIM_ * 2);
  unsigned short* wqbT = (unsigned short*)alloc((size_t)DIM_ * DIM_ * 2);
  unsigned short* wkbT = (unsigned short*)alloc((size_t)DIM_ * DIM_ * 2);
  unsigned short* wvbT = (unsigned short*)alloc((size_t)DIM_ * DIM_ * 2);
  unsigned short* wobT = (unsigned short*)alloc((size_t)DIM_ * DIM_ * 2);
  float* ctx_num = (float*)alloc((size_t)MB_ * HEADS_ * DH_ * DH_ * 4);
  float* denomG = (float*)alloc((size_t)MB_ * DIM_ * 4);
  unsigned short* ctxbfT = (unsigned short*)alloc((size_t)MB_ * HEADS_ * DH_ * DH_ * 2);

  // Accumulators must be re-zeroed every call (graph replay).
  k_zero<<<512, 256, 0, stream>>>(ctx_num, MB_ * HEADS_ * DH_ * DH_);
  k_zero<<<64, 256, 0, stream>>>(denomG, MB_ * DIM_);
  // next_mem = hiddens (concat(smem,hiddens)[-2048:]).
  k_copy4<<<2048, 256, 0, stream>>>((const float4*)hiddens, (float4*)nmem,
                                    (int)(NMEM_ELEMS / 4));
  k_wconv<<<512, 256, 0, stream>>>(w_q, w_kv, w_out, wqbT, wkbT, wvbT, wobT);
  k_layernorm<<<QROWS_ / 8, 256, 0, stream>>>(lmem, normed);
  k_context<<<dim3(17, 48), 256, 0, stream>>>(smem, hiddens, normed, wkbT, wvbT,
                                              ctx_num, denomG);
  k_ctxfin<<<256, 256, 0, stream>>>(ctx_num, denomG, ctxbfT);
  k_qpath<<<QROWS_ / 32, 256, 0, stream>>>(lmem, normed, wqbT, wobT, ctxbfT, rezero, nlm);
}
